// AdderConv_77034533421675
// MI455X (gfx1250) — compile-verified, matrix operand order checked by reading, not run on hardware
//
#include <hip/hip_runtime.h>

typedef __attribute__((ext_vector_type(16))) _Float16 v16h;
typedef __attribute__((ext_vector_type(8)))  float    v8f;

#define BATCH 128

// ---------------------------------------------------------------------------
// AdderNet layer: out[b,o,h,w] = relu( bias[o] - scale[o] * sum_{c,kh,kw} |x - w| )
// 3x3 kernel, stride 1, zero-pad 1 (padding contributes |0 - w| terms).
// One block per (batch, out-channel); weight slice cached in LDS.
// ---------------------------------------------------------------------------
__global__ void adder_abr_kernel(const float* __restrict__ x,
                                 const float* __restrict__ w,
                                 const float* __restrict__ scale,
                                 const float* __restrict__ bias,
                                 float* __restrict__ out,
                                 int C, int H, int W) {
  extern __shared__ float wl[];          // C*9 floats (<= 4608 bytes)
  const int b  = blockIdx.x;
  const int o  = blockIdx.y;
  const int O  = gridDim.y;
  const int nw = C * 9;
  for (int i = threadIdx.x; i < nw; i += blockDim.x) wl[i] = w[o * nw + i];
  __syncthreads();

  const float sc = scale[o];
  const float bi = bias[o];
  const int   HW = H * W;
  const float* __restrict__ xb = x + (size_t)b * C * HW;
  float* __restrict__ ob = out + ((size_t)b * O + o) * HW;

  for (int p = threadIdx.x; p < HW; p += blockDim.x) {
    const int h  = p / W;
    const int w0 = p - h * W;
    float acc = 0.f;
    for (int c = 0; c < C; ++c) {
      const float* __restrict__ xc = xb + c * HW;
      const float* __restrict__ wc = wl + c * 9;
#pragma unroll
      for (int kh = 0; kh < 3; ++kh) {
        const int  hh = h + kh - 1;
        const bool hv = ((unsigned)hh < (unsigned)H);
        const float* __restrict__ xr = xc + hh * W;
#pragma unroll
        for (int kw = 0; kw < 3; ++kw) {
          const int   ww = w0 + kw - 1;
          const float xv = (hv && ((unsigned)ww < (unsigned)W)) ? xr[ww] : 0.f;
          acc += fabsf(xv - wc[kh * 3 + kw]);
        }
      }
    }
    ob[p] = fmaxf(0.f, bi - acc * sc);   // relu((-acc)*s + b)
  }
}

// ---------------------------------------------------------------------------
// MaxPool 2x2 stride 2
// ---------------------------------------------------------------------------
__global__ void maxpool2_kernel(const float* __restrict__ in,
                                float* __restrict__ out,
                                int C, int H, int W, int total) {
  int idx = blockIdx.x * blockDim.x + threadIdx.x;
  if (idx >= total) return;
  const int Wo = W >> 1, Ho = H >> 1;
  int wo = idx % Wo; int t = idx / Wo;
  int ho = t % Ho;   t /= Ho;
  int c  = t % C;    int b = t / C;
  const float* p = in + (((size_t)b * C + c) * H + ho * 2) * W + wo * 2;
  out[idx] = fmaxf(fmaxf(p[0], p[1]), fmaxf(p[W], p[W + 1]));
}

// ---------------------------------------------------------------------------
// f32 -> f16 converters for the WMMA FC
// ---------------------------------------------------------------------------
__global__ void cvt_x_kernel(const float* __restrict__ in, _Float16* __restrict__ o, int n) {
  int i = blockIdx.x * blockDim.x + threadIdx.x;
  if (i < n) o[i] = (_Float16)in[i];
}

// fc_w is [10,2048]; pad to [16,2048] with zeros so one WMMA N-tile covers it.
__global__ void cvt_w_kernel(const float* __restrict__ in, _Float16* __restrict__ o) {
  int i = blockIdx.x * blockDim.x + threadIdx.x;
  if (i < 16 * 2048)
    o[i] = (i < 10 * 2048) ? (_Float16)in[i] : (_Float16)0.f;
}

// ---------------------------------------------------------------------------
// FC: out[128,10] = X[128,2048] @ W[10,2048]^T + b  via v_wmma_f32_16x16x32_f16.
// One wave (32 threads) per 16-row batch tile; 64 K-steps of 32.
// A layout: lane -> M (=lane&15); K pairs per VGPR, lanes>=16 offset K by 8,
//           VGPRs 4..7 hold K+16.  B mirrors A with lane -> N.
// C/D layout: VGPR r -> M = r + 8*(lane>=16), N = lane&15.
// ---------------------------------------------------------------------------
__global__ void fc_wmma_kernel(const _Float16* __restrict__ X,   // [128][2048]
                               const _Float16* __restrict__ Wm,  // [16][2048]
                               const float* __restrict__ fcb,    // [10]
                               float* __restrict__ out) {        // [128][10]
  const int tile  = blockIdx.x;        // 0..7
  const int lane  = threadIdx.x;       // 0..31
  const int l     = lane & 15;
  const int half  = lane >> 4;
  const int kbase = half * 8;

  const _Float16* __restrict__ xrow = X  + (size_t)(tile * 16 + l) * 2048;
  const _Float16* __restrict__ wrow = Wm + (size_t)l * 2048;

  v8f acc = {};
  for (int k0 = 0; k0 < 2048; k0 += 32) {
    v16h a, bm;
#pragma unroll
    for (int i = 0; i < 8; ++i) {
      a[i]      = xrow[k0 + kbase + i];
      a[8 + i]  = xrow[k0 + 16 + kbase + i];
      bm[i]     = wrow[k0 + kbase + i];
      bm[8 + i] = wrow[k0 + 16 + kbase + i];
    }
    acc = __builtin_amdgcn_wmma_f32_16x16x32_f16(
        /*neg_a=*/false, a, /*neg_b=*/false, bm,
        /*c_mod=*/(short)0, acc, /*reuse_a=*/false, /*reuse_b=*/false);
  }

  const int n = l;
  if (n < 10) {
    const float bv = fcb[n];
#pragma unroll
    for (int r = 0; r < 8; ++r) {
      const int m = tile * 16 + r + 8 * half;
      out[m * 10 + n] = acc[r] + bv;
    }
  }
}

// ---------------------------------------------------------------------------
extern "C" void kernel_launch(void* const* d_in, const int* in_sizes, int n_in,
                              void* d_out, int out_size, void* d_ws, size_t ws_size,
                              hipStream_t stream) {
  (void)in_sizes; (void)n_in; (void)out_size; (void)ws_size;

  // setup_inputs dict order: x, (w_i, s_i, b_i) x 8, fc_w, fc_b
  const float* x = (const float*)d_in[0];
  const float *Wt[8], *St[8], *Bt[8];
  for (int i = 0; i < 8; ++i) {
    Wt[i] = (const float*)d_in[1 + 3 * i];
    St[i] = (const float*)d_in[2 + 3 * i];
    Bt[i] = (const float*)d_in[3 + 3 * i];
  }
  const float* fcw = (const float*)d_in[25];
  const float* fcb = (const float*)d_in[26];
  float* out = (float*)d_out;

  char* ws = (char*)d_ws;
  float*    bufA = (float*)(ws);                              // 16 MB
  float*    bufB = (float*)(ws + (size_t)(16u << 20));        // 16 MB
  _Float16* X16  = (_Float16*)(ws + (size_t)(32u << 20));     // 512 KB
  _Float16* W16  = (_Float16*)(ws + (size_t)(32u << 20) + (1u << 20)); // 64 KB

  auto adder = [&](const float* in, int li, float* o, int C, int O, int H, int W) {
    int hw = H * W;
    int threads = (hw >= 256) ? 256 : hw;
    dim3 grid(BATCH, O);
    size_t shmem = (size_t)C * 9 * sizeof(float);
    adder_abr_kernel<<<grid, threads, shmem, stream>>>(in, Wt[li], St[li], Bt[li], o, C, H, W);
  };
  auto pool = [&](const float* in, float* o, int C, int H, int W) {
    int total = BATCH * C * (H / 2) * (W / 2);
    maxpool2_kernel<<<(total + 255) / 256, 256, 0, stream>>>(in, o, C, H, W, total);
  };

  adder(x,    0, bufA,   3,  32, 32, 32);   // L1: [128,32,32,32]
  adder(bufA, 1, bufB,  32,  32, 32, 32);   // L2
  pool (bufB,    bufA,  32, 32, 32);        // -> [128,32,16,16]
  adder(bufA, 2, bufB,  32,  64, 16, 16);   // L3
  adder(bufB, 3, bufA,  64,  64, 16, 16);   // L4
  pool (bufA,    bufB,  64, 16, 16);        // -> [128,64,8,8]
  adder(bufB, 4, bufA,  64, 128,  8,  8);   // L5
  adder(bufA, 5, bufB, 128, 128,  8,  8);   // L6
  adder(bufB, 6, bufA, 128, 128,  8,  8);   // L7
  adder(bufA, 7, bufB, 128, 128,  8,  8);   // L8
  pool (bufB,    bufA, 128,  8,  8);        // -> [128,128,4,4] == [128,2048]

  // FC via WMMA
  {
    int nx = BATCH * 2048;
    cvt_x_kernel<<<(nx + 255) / 256, 256, 0, stream>>>(bufA, X16, nx);
    int nw = 16 * 2048;
    cvt_w_kernel<<<(nw + 255) / 256, 256, 0, stream>>>(fcw, W16);
    fc_wmma_kernel<<<BATCH / 16, 32, 0, stream>>>(X16, W16, fcb, out);
  }
}
